// UnfoldPOCS_2826088481273
// MI455X (gfx1250) — matrix operand order repaired and checked by reading, not compiled, not run on hardware
//
#include <hip/hip_runtime.h>

// ---------------- problem constants ----------------
#define BATCH 2048
#define KD    128
#define MD    640           // 5*KD
#define ALPHA (1.0f / 128.0f)
#define VARTHETA 0.1f
#define PMAX  10.0f
#define BIGV  1000000000.0f

// ---------------- leading dims ----------------
#define LDH   136  // bf16 ld for 128x128 HW (row-major, A operand)
#define LD32  40   // bf16 ld for 128x32 row-major (A operands)
#define LDG   20   // f32  ld for 128x16 fp32 results
#define LDP16 20   // dword ld for packed-B 16-col arrays (8*20 % 64 == 32 -> no conflicts)
#define LDP32 36   // dword ld for packed-B 32-col arrays (8*36 % 64 == 32 -> no conflicts)

// ---------------- LDS layout (byte offsets) ----------------
#define OFF_HW      0        // ushort 128*136 = 34816
#define OFF_X       34816    // float  640     =  2560
#define OFF_X1      37376    // ushort 128*40  = 10240  (X1/X2 row-major, A)
#define OFF_SX1     47616    // ushort 128*40  = 10240  (SX1 row-major, A)
#define OFF_Z1      57856    // ushort 128*40  = 10240  (Z1/Z2 row-major, A)
#define OFF_G1      68096    // float  128*20  = 10240  (g1 out / tilde)
#define OFF_MISC    78336    // float  16      =    64
#define OFF_X1P     78400    // uint   64*20   =  5120  (X1 packed-B)
#define OFF_UP      83520    // uint   64*36   =  9216  (U packed-B)
#define OFF_V1P     92736    // uint   64*20   =  5120  (V1/V2 packed-B)
#define OFF_W0A1P   97856    // uint   16*36   =  2304
#define OFF_W1A1P   100160   // uint   16*36   =  2304
#define OFF_W0B1P   102464   // uint   16*20   =  1280
#define OFF_W1B1P   103744   // uint   16*20   =  1280
#define OFF_W0A2MP  105024   // uint   64*36   =  9216
#define OFF_W1A2MP  114240   // uint   64*36   =  9216
#define OFF_W0A2TP  123456   // uint   16*36   =  2304
#define OFF_W1A2TP  125760   // uint   16*36   =  2304
#define OFF_W0B2P   128064   // uint   16*20   =  1280
#define OFF_W1B2P   129344   // uint   16*20   =  1280
#define SMEM_BYTES  130624

// ---------------- WMMA plumbing (gfx1250, wave32) ----------------
typedef __attribute__((ext_vector_type(16))) __bf16 v16bf;
typedef __attribute__((ext_vector_type(8)))  float  v8f;

struct Frag { unsigned int u[8]; };

static __device__ __forceinline__ v16bf frag_bits(const Frag& f) {
  return __builtin_bit_cast(v16bf, f);
}

// hardware fptrunc (RNE) instead of software bit-twiddled rounding
static __device__ __forceinline__ unsigned short f2bf(float x) {
  return __builtin_bit_cast(unsigned short, (__bf16)x);
}
static __device__ __forceinline__ unsigned int f2bf_pk(float lo, float hi) {
  __bf16 a = (__bf16)lo, b = (__bf16)hi;
  return (unsigned int)__builtin_bit_cast(unsigned short, a) |
         ((unsigned int)__builtin_bit_cast(unsigned short, b) << 16);
}

// A fragment: 16x32 bf16 tile from row-major LDS (ISA 7.12.2 16-bit A layout).
static __device__ __forceinline__ Frag load_a(const unsigned short* base, int ld,
                                              int row0, int col0, int lane) {
  Frag f;
  const int m = lane & 15, half = lane >> 4;
  const unsigned short* p = base + (row0 + m) * ld + col0;
#pragma unroll
  for (int v = 0; v < 8; ++v) {
    const int k = (v >> 2) * 16 + half * 8 + (v & 3) * 2;
    f.u[v] = *(const unsigned int*)(p + k);   // packed bf16 pair (K even)
  }
  return f;
}

// B fragment from K-pair-packed LDS: dword (pr, n) = bf16{K=2pr, K=2pr+1} of column n.
// Lane layout: lane = column (n, +16 -> K half), VGPR v holds K pair ((v>>2)*16+half*8+(v&3)*2, +1).
static __device__ __forceinline__ Frag load_bp(const unsigned int* base, int ldp,
                                               int k0, int col0, int lane) {
  Frag f;
  const int n = lane & 15, half = lane >> 4;
  const unsigned int* p = base + ((k0 >> 1) + half * 4) * ldp + col0 + n;
#pragma unroll
  for (int v = 0; v < 8; ++v) f.u[v] = p[((v >> 2) * 8 + (v & 3)) * ldp];
  return f;
}

static __device__ __forceinline__ v8f wmma_bf16(const Frag& a, const Frag& b, v8f c) {
  return __builtin_amdgcn_wmma_f32_16x16x32_bf16(false, frag_bits(a),
                                                 false, frag_bits(b),
                                                 (short)0, c, false, false);
}

static __device__ __forceinline__ v8f zero8() {
  v8f c;
#pragma unroll
  for (int i = 0; i < 8; ++i) c[i] = 0.0f;
  return c;
}

// C/D: VGPR v -> row half*8+v, col = lane&15 (ISA 7.12.2 32-bit C/D 16x16).
static __device__ __forceinline__ void store_c_f32(float* base, int ld, int row0,
                                                   int col0, int lane, v8f c) {
  const int n = lane & 15, half = lane >> 4;
#pragma unroll
  for (int v = 0; v < 8; ++v)
    base[(row0 + half * 8 + v) * ld + col0 + n] = c[v];
}

static __device__ __forceinline__ void store_c_bf16(unsigned short* base, int ld,
                                                    int row0, int col0, int lane,
                                                    v8f c, bool relu) {
  const int n = lane & 15, half = lane >> 4;
#pragma unroll
  for (int v = 0; v < 8; ++v) {
    float x = c[v];
    if (relu) x = fmaxf(x, 0.0f);
    base[(row0 + half * 8 + v) * ld + col0 + n] = f2bf(x);
  }
}

// Store accumulator directly in packed-B layout: rows (v, v+1) pack into one dword.
static __device__ __forceinline__ void store_c_packed(unsigned int* base, int ldp,
                                                      int row0, int col0, int lane,
                                                      v8f c) {
  const int n = lane & 15, half = lane >> 4;
#pragma unroll
  for (int v = 0; v < 8; v += 2) {
    base[((row0 >> 1) + half * 4 + (v >> 1)) * ldp + col0 + n] = f2bf_pk(c[v], c[v + 1]);
  }
}

// fp32 weight [rows x cols] row-major -> zero-padded packed-B LDS [totPR x ldp] dwords.
static __device__ __forceinline__ void load_w_packed(unsigned int* dst, const float* src,
                                                     int rows, int cols, int ldp,
                                                     int totPR, int tid) {
  for (int i = tid; i < totPR * ldp; i += 256) {
    const int pr = i / ldp, c = i - pr * ldp;
    const int r0 = 2 * pr, r1 = r0 + 1;
    const float lo = (r0 < rows && c < cols) ? src[r0 * cols + c] : 0.0f;
    const float hi = (r1 < rows && c < cols) ? src[r1 * cols + c] : 0.0f;
    dst[i] = f2bf_pk(lo, hi);
  }
}

// ---------------- the fused per-batch kernel ----------------
__global__ __launch_bounds__(256)
void unfold_pocs_kernel(const int* __restrict__ num_itr, const int* __restrict__ kth,
                        const float* __restrict__ HW,  const float* __restrict__ x0,
                        const float* __restrict__ nu3p, const float* __restrict__ tgp,
                        const float* __restrict__ g1W0a, const float* __restrict__ g1W1a,
                        const float* __restrict__ g1W0b, const float* __restrict__ g1W1b,
                        const float* __restrict__ g2W0a, const float* __restrict__ g2W1a,
                        const float* __restrict__ g2W0b, const float* __restrict__ g2W1b,
                        float* __restrict__ out) {
  extern __shared__ unsigned char smem[];
  unsigned short* sHW    = (unsigned short*)(smem + OFF_HW);
  float*          sX     = (float*)(smem + OFF_X);
  unsigned short* sX1    = (unsigned short*)(smem + OFF_X1);
  unsigned short* sSX1   = (unsigned short*)(smem + OFF_SX1);
  unsigned short* sZ1    = (unsigned short*)(smem + OFF_Z1);
  float*          sG1    = (float*)(smem + OFF_G1);
  float*          sMisc  = (float*)(smem + OFF_MISC);
  unsigned int*   sX1p   = (unsigned int*)(smem + OFF_X1P);
  unsigned int*   sUp    = (unsigned int*)(smem + OFF_UP);
  unsigned int*   sV1p   = (unsigned int*)(smem + OFF_V1P);
  unsigned int*   sW0a1p  = (unsigned int*)(smem + OFF_W0A1P);
  unsigned int*   sW1a1p  = (unsigned int*)(smem + OFF_W1A1P);
  unsigned int*   sW0b1p  = (unsigned int*)(smem + OFF_W0B1P);
  unsigned int*   sW1b1p  = (unsigned int*)(smem + OFF_W1B1P);
  unsigned int*   sW0a2mp = (unsigned int*)(smem + OFF_W0A2MP);
  unsigned int*   sW1a2mp = (unsigned int*)(smem + OFF_W1A2MP);
  unsigned int*   sW0a2tp = (unsigned int*)(smem + OFF_W0A2TP);
  unsigned int*   sW1a2tp = (unsigned int*)(smem + OFF_W1A2TP);
  unsigned int*   sW0b2p  = (unsigned int*)(smem + OFF_W0B2P);
  unsigned int*   sW1b2p  = (unsigned int*)(smem + OFF_W1B2P);

  const int tid  = threadIdx.x;
  const int lane = tid & 31;
  const int wave = tid >> 5;
  const int b    = blockIdx.x;

  const bool isLast = (kth[0] == num_itr[0] - 1);
  const float nu3 = nu3p[0];
  const float tg  = tgp[0];

  const float* HWb = HW + (size_t)b * (KD * KD);
  const float* xb  = x0 + (size_t)b * MD;

  // ---- Stage 0: HW[b] fp32->bf16, x row, packed weights, zero pads
  for (int i = tid; i < KD * KD; i += 256)
    sHW[(i >> 7) * LDH + (i & 127)] = f2bf(HWb[i]);
  for (int i = tid; i < MD; i += 256) sX[i] = xb[i];
  for (int i = tid; i < KD * LD32; i += 256) { sX1[i] = 0; sSX1[i] = 0; }
  for (int i = tid; i < 64 * LDP16; i += 256) sX1p[i] = 0;
  load_w_packed(sW0a1p, g1W0a, 5, 32, LDP32, 16, tid);
  load_w_packed(sW1a1p, g1W1a, 5, 32, LDP32, 16, tid);
  load_w_packed(sW0b1p, g1W0b, 32, 2, LDP16, 16, tid);
  load_w_packed(sW1b1p, g1W1b, 32, 2, LDP16, 16, tid);
  load_w_packed(sW0a2mp, g2W0a, 128, 32, LDP32, 64, tid);
  load_w_packed(sW1a2mp, g2W1a, 128, 32, LDP32, 64, tid);
  load_w_packed(sW0a2tp, g2W0a + 128 * 32, 5, 32, LDP32, 16, tid);
  load_w_packed(sW1a2tp, g2W1a + 128 * 32, 5, 32, LDP32, 16, tid);
  load_w_packed(sW0b2p, g2W0b, 32, 5, LDP16, 16, tid);
  load_w_packed(sW1b2p, g2W1b, 32, 5, LDP16, 16, tid);
  __syncthreads();

  // ---- Stage 1 (GNN 1): X1[k][f] = x[f*K+k], row-major (A) + packed (B)
  for (int i = tid; i < MD; i += 256)
    sX1[(i & 127) * LD32 + (i >> 7)] = f2bf(sX[i]);
  for (int i = tid; i < 64 * 5; i += 256) {
    const int pr = i & 63, f = i >> 6;          // packed row, feature
    sX1p[pr * LDP16 + f] = f2bf_pk(sX[f * KD + 2 * pr], sX[f * KD + 2 * pr + 1]);
  }
  __syncthreads();

  // SX1 = HW @ X1  (8 row-tiles -> one per wave)
  {
    const int m0 = wave * 16;
    v8f c = zero8();
    for (int k0 = 0; k0 < KD; k0 += 32)
      c = wmma_bf16(load_a(sHW, LDH, m0, k0, lane), load_bp(sX1p, LDP16, k0, 0, lane), c);
    store_c_bf16(sSX1, LD32, m0, 0, lane, c, false);
  }
  __syncthreads();

  // Z1 = relu(X1 @ W0a1 + SX1 @ W1a1)
  for (int t = wave; t < 16; t += 8) {
    const int m0 = (t >> 1) * 16, n0 = (t & 1) * 16;
    v8f c = zero8();
    c = wmma_bf16(load_a(sX1, LD32, m0, 0, lane), load_bp(sW0a1p, LDP32, 0, n0, lane), c);
    c = wmma_bf16(load_a(sSX1, LD32, m0, 0, lane), load_bp(sW1a1p, LDP32, 0, n0, lane), c);
    store_c_bf16(sZ1, LD32, m0, n0, lane, c, true);
  }
  __syncthreads();

  // V1 = Z1 @ W1b1   ( (HW@Z1)@W1b1 == HW@(Z1@W1b1) ), stored packed-B
  {
    const int m0 = wave * 16;
    v8f c = zero8();
    c = wmma_bf16(load_a(sZ1, LD32, m0, 0, lane), load_bp(sW1b1p, LDP16, 0, 0, lane), c);
    store_c_packed(sV1p, LDP16, m0, 0, lane, c);
  }
  __syncthreads();

  // G1 = Z1 @ W0b1 + HW @ V1   (fp32 out, cols 0..1 valid)
  {
    const int m0 = wave * 16;
    v8f c = zero8();
    c = wmma_bf16(load_a(sZ1, LD32, m0, 0, lane), load_bp(sW0b1p, LDP16, 0, 0, lane), c);
    for (int k0 = 0; k0 < KD; k0 += 32)
      c = wmma_bf16(load_a(sHW, LDH, m0, k0, lane), load_bp(sV1p, LDP16, k0, 0, lane), c);
    store_c_f32(sG1, LDG, m0, 0, lane, c);
  }
  __syncthreads();

  // lbd = relu(mean over rows)
  if (tid < 2) {
    float s = 0.0f;
    for (int k2 = 0; k2 < KD; ++k2) s += sG1[k2 * LDG + tid];
    sMisc[tid] = fmaxf(s * (1.0f / KD), 0.0f);
  }
  __syncthreads();

  // s1 -= lbd0 * ALPHA/(1+s1);  s4 += lbd1 * ALPHA*VARTHETA
  if (tid < KD) {
    const float s1 = sX[KD + tid];
    sX[KD + tid] = s1 - sMisc[0] * (ALPHA / (1.0f + s1));
  } else if (tid < 2 * KD) {
    sX[4 * KD + (tid - KD)] += sMisc[1] * (ALPHA * VARTHETA);
  }
  __syncthreads();

  // ---- Stage 2 (GNN 2 on Xin=[HW|X2]), reassociated to avoid HW@HW
  for (int i = tid; i < MD; i += 256)
    sX1[(i & 127) * LD32 + (i >> 7)] = f2bf(sX[i]);     // X2 (A operand only)
  __syncthreads();

  // U = HW @ W1a2m + X2 @ W1a2t   ( == Xin @ W1a2 ), stored packed-B
  for (int t = wave; t < 16; t += 8) {
    const int m0 = (t >> 1) * 16, n0 = (t & 1) * 16;
    v8f c = zero8();
    for (int k0 = 0; k0 < KD; k0 += 32)
      c = wmma_bf16(load_a(sHW, LDH, m0, k0, lane), load_bp(sW1a2mp, LDP32, k0, n0, lane), c);
    c = wmma_bf16(load_a(sX1, LD32, m0, 0, lane), load_bp(sW1a2tp, LDP32, 0, n0, lane), c);
    store_c_packed(sUp, LDP32, m0, n0, lane, c);
  }
  __syncthreads();

  // Z2 = relu(HW@W0a2m + X2@W0a2t + HW@U)   ( == relu(Xin@W0a2 + (HW@Xin)@W1a2) )
  for (int t = wave; t < 16; t += 8) {
    const int m0 = (t >> 1) * 16, n0 = (t & 1) * 16;
    v8f c = zero8();
    for (int k0 = 0; k0 < KD; k0 += 32)
      c = wmma_bf16(load_a(sHW, LDH, m0, k0, lane), load_bp(sW0a2mp, LDP32, k0, n0, lane), c);
    c = wmma_bf16(load_a(sX1, LD32, m0, 0, lane), load_bp(sW0a2tp, LDP32, 0, n0, lane), c);
    for (int k0 = 0; k0 < KD; k0 += 32)
      c = wmma_bf16(load_a(sHW, LDH, m0, k0, lane), load_bp(sUp, LDP32, k0, n0, lane), c);
    store_c_bf16(sZ1, LD32, m0, n0, lane, c, true);
  }
  __syncthreads();

  // V2 = Z2 @ W1b2, stored packed-B
  {
    const int m0 = wave * 16;
    v8f c = zero8();
    c = wmma_bf16(load_a(sZ1, LD32, m0, 0, lane), load_bp(sW1b2p, LDP16, 0, 0, lane), c);
    store_c_packed(sV1p, LDP16, m0, 0, lane, c);
  }
  __syncthreads();

  // tilde = Z2 @ W0b2 + HW @ V2   (fp32, cols 0..4 valid)
  {
    const int m0 = wave * 16;
    v8f c = zero8();
    c = wmma_bf16(load_a(sZ1, LD32, m0, 0, lane), load_bp(sW0b2p, LDP16, 0, 0, lane), c);
    for (int k0 = 0; k0 < KD; k0 += 32)
      c = wmma_bf16(load_a(sHW, LDH, m0, k0, lane), load_bp(sV1p, LDP16, k0, 0, lane), c);
    store_c_f32(sG1, LDG, m0, 0, lane, c);
  }
  __syncthreads();

  // x += tilde^T flattened: x[f*K+k] += tilde[k][f]
  for (int i = tid; i < MD; i += 256)
    sX[i] += sG1[(i & 127) * LDG + (i >> 7)];
  __syncthreads();

  // ---- POCS projection of power band
  if (tid == 0) {
    float s = 0.0f;
    for (int i = 0; i < KD; ++i) s += sX[i];
    sMisc[2] = s - PMAX;
  }
  __syncthreads();
  {
    const float fp = sMisc[2];
    if (fp > 0.0f && tid < KD) sX[tid] -= fp * (1.0f / KD);
  }
  __syncthreads();

  // ---- band clamps (+ final relu) and write x
  {
    const float ivn = 1.0f / (1.0f + nu3);
    const float Vnu = 1.0f - ivn * ivn;
    const float ivg = 1.0f / (1.0f + tg);
    const float Vg  = 1.0f - ivg * ivg;
    for (int i = tid; i < MD; i += 256) {
      float v = sX[i];
      const int band = i >> 7;
      const float lo = (band == 1) ? nu3 : ((band == 3) ? Vnu : 0.0f);
      const float hi = (band == 2) ? tg : ((band == 3) ? Vg : BIGV);
      v = fmaxf(v, lo);
      v = fminf(v, hi);
      if (isLast) v = fmaxf(v, 0.0f);
      sX[i] = v;
      out[(size_t)b * MD + i] = v;
    }
  }
  __syncthreads();

  // ---- final residuals f5,f6,f4,f2 (exact fp32 HW re-read, coalesced across lanes)
  if (isLast && tid < KD) {
    const int j = tid;
    const float p  = sX[j];
    const float nu = sX[KD + j];
    const float xg = sX[2 * KD + j];
    const float xv = sX[3 * KD + j];
    const float x4 = sX[4 * KD + j];
    const float iv = 1.0f / (1.0f + xg);
    const float f5 = 1.0f - iv * iv - xv;
    const float f6 = sqrtf(xv) - x4;
    float acc = 0.0f;
    for (int i2 = 0; i2 < KD; ++i2)
      acc = fmaf(sX[i2], HWb[i2 * KD + j], acc);       // (p . HW)[j]
    const float diag = HWb[j * KD + j];
    const float top  = p * diag;
    const float down = acc - top + 1.0f;
    const float f4  = top - down * xg;
    const float f2v = nu * down - top;
    const size_t base = (size_t)BATCH * MD;
    const size_t bk   = (size_t)BATCH * KD;
    out[base + 0 * bk + (size_t)b * KD + j] = f5;
    out[base + 1 * bk + (size_t)b * KD + j] = f6;
    out[base + 2 * bk + (size_t)b * KD + j] = f4;
    out[base + 3 * bk + (size_t)b * KD + j] = f2v;
  }
}

// ---------------- launch ----------------
extern "C" void kernel_launch(void* const* d_in, const int* in_sizes, int n_in,
                              void* d_out, int out_size, void* d_ws, size_t ws_size,
                              hipStream_t stream) {
  (void)in_sizes; (void)n_in; (void)out_size; (void)d_ws; (void)ws_size;
  const int*   num_itr = (const int*)d_in[0];
  const int*   kth     = (const int*)d_in[1];
  const float* HW      = (const float*)d_in[2];
  const float* x0      = (const float*)d_in[3];
  const float* nu3     = (const float*)d_in[4];
  const float* tg      = (const float*)d_in[5];
  // d_in[6..11] = A9,A1,A3,A7,b7,A10 : unused by the reference
  const float* g1W0a = (const float*)d_in[12];
  const float* g1W1a = (const float*)d_in[13];
  const float* g1W0b = (const float*)d_in[14];
  const float* g1W1b = (const float*)d_in[15];
  const float* g2W0a = (const float*)d_in[16];
  const float* g2W1a = (const float*)d_in[17];
  const float* g2W0b = (const float*)d_in[18];
  const float* g2W1b = (const float*)d_in[19];
  float* out = (float*)d_out;

  // allow >64KB dynamic LDS (host-side attr; not a stream op, graph-capture safe)
  hipFuncSetAttribute((const void*)unfold_pocs_kernel,
                      hipFuncAttributeMaxDynamicSharedMemorySize, SMEM_BYTES);

  unfold_pocs_kernel<<<dim3(BATCH), dim3(256), SMEM_BYTES, stream>>>(
      num_itr, kth, HW, x0, nu3, tg,
      g1W0a, g1W1a, g1W0b, g1W1b,
      g2W0a, g2W1a, g2W0b, g2W1b, out);
}
